// VoiceKeyModel_17609365914344
// MI455X (gfx1250) — matrix operand order repaired and checked by reading, not compiled, not run on hardware
//
#include <hip/hip_runtime.h>

// ---------------------------------------------------------------------------
// CDNA5 (gfx1250) wave32 WMMA implementation of the VoiceKey model.
// All big GEMMs run on v_wmma_f32_16x16x32_f16 (f16 in, f32 accumulate).
// Conv K-orderings are chosen so B fragments are contiguous aligned LDS runs:
//   conv2: kk = k*64  + ic  (y1 kept transposed as y1t[p][ic])
//   conv3: kk = k*128 + ic  (y2 kept transposed as y2t[p][oc])
// ---------------------------------------------------------------------------

typedef __attribute__((ext_vector_type(16))) _Float16 v16h;
typedef __attribute__((ext_vector_type(8)))  float    v8f;
typedef __attribute__((ext_vector_type(4)))  unsigned int u32x4;

union FragH {
    v16h h;
    u32x4 q[2];
    unsigned int w[8];
};

__device__ __forceinline__ v8f wmma16(v16h a, v16h b, v8f c) {
    // D = A(16x32 f16) * B(32x16 f16) + C(16x16 f32)
    return __builtin_amdgcn_wmma_f32_16x16x32_f16(false, a, false, b,
                                                  (short)0, c, false, false);
}

__device__ __forceinline__ v8f zero8() {
    v8f z = {0.f, 0.f, 0.f, 0.f, 0.f, 0.f, 0.f, 0.f};
    return z;
}

__device__ __forceinline__ v16h frag_zero() {
    FragH f;
#pragma unroll
    for (int i = 0; i < 8; ++i) f.w[i] = 0u;
    return f.h;
}

// A fragment (16x32, f16) from a row-major matrix (contiguous K order):
// lanes 0-15 : row = lane,    halves 0..15 = K k0..k0+15
// lanes16-31 : row = lane-16, halves 0..15 = K k0+16..k0+31
__device__ __forceinline__ v16h load_a_frag(const _Float16* base, int ld,
                                            int row, int k0, int lane) {
    const _Float16* p = base + row * ld + k0 + ((lane >> 4) << 4);
    FragH f;
    f.q[0] = *(const u32x4*)p;
    f.q[1] = *(const u32x4*)(p + 8);
    return f.h;
}

// B fragment (32x16, f16) from K-contiguous (transposed) storage Bt[col][k].
// lanes 0-15: col = lane, K = k0..k0+15 ; lanes 16-31: col = lane-16, K = k0+16..+31
__device__ __forceinline__ v16h load_b_frag(const _Float16* base, int ld,
                                            int col, int k0, int lane) {
    const _Float16* p = base + col * ld + k0 + ((lane >> 4) << 4);
    FragH f;
    f.q[0] = *(const u32x4*)p;
    f.q[1] = *(const u32x4*)(p + 8);
    return f.h;
}

// B fragment whose 16 K-values sit at a single contiguous 16B-aligned address.
__device__ __forceinline__ v16h load_b_run(const _Float16* p) {
    FragH f;
    f.q[0] = *(const u32x4*)p;
    f.q[1] = *(const u32x4*)(p + 8);
    return f.h;
}

// SiLU with fast reciprocal (v_rcp_f32): result is rounded to f16 right
// after, so the ~1 ulp rcp error is invisible.
__device__ __forceinline__ float silu(float x) {
    return x * __builtin_amdgcn_rcpf(1.f + __expf(-x));
}

// ------------------------------- prep kernels ------------------------------

__global__ void k_fold_conv1(const float* __restrict__ w, const float* __restrict__ b,
                             const float* __restrict__ g, const float* __restrict__ beta,
                             const float* __restrict__ m, const float* __restrict__ v,
                             float* __restrict__ wf, float* __restrict__ bf) {
    int oc = threadIdx.x;  // 64 threads
    float scale = g[oc] * rsqrtf(v[oc] + 1e-5f);
    for (int k = 0; k < 26; ++k) wf[oc * 26 + k] = w[oc * 26 + k] * scale;
    bf[oc] = (b[oc] - m[oc]) * scale + beta[oc];
}

// conv2 weights: fold BN, f16, reorder (oc, ic*26+k) -> (oc, k*64+ic)
__global__ void k_fold_conv2(const float* __restrict__ w, const float* __restrict__ b,
                             const float* __restrict__ g, const float* __restrict__ beta,
                             const float* __restrict__ m, const float* __restrict__ v,
                             _Float16* __restrict__ wh, float* __restrict__ bf) {
    int oc = blockIdx.x;  // 128
    float scale = g[oc] * rsqrtf(v[oc] + 1e-5f);
    for (int i = threadIdx.x; i < 1664; i += blockDim.x) {
        int ic = i / 26, k = i - ic * 26;
        wh[oc * 1664 + k * 64 + ic] = (_Float16)(w[oc * 1664 + i] * scale);
    }
    if (threadIdx.x == 0) bf[oc] = (b[oc] - m[oc]) * scale + beta[oc];
}

// conv3 weights: fold BN, f16, reorder (oc, ic*25+k) -> (oc, k*128+ic)
__global__ void k_fold_conv3(const float* __restrict__ w, const float* __restrict__ b,
                             const float* __restrict__ g, const float* __restrict__ beta,
                             const float* __restrict__ m, const float* __restrict__ v,
                             _Float16* __restrict__ wh, float* __restrict__ bf) {
    int oc = blockIdx.x;  // 256
    float scale = g[oc] * rsqrtf(v[oc] + 1e-5f);
    for (int i = threadIdx.x; i < 3200; i += blockDim.x) {
        int ic = i / 25, k = i - ic * 25;
        wh[oc * 3200 + k * 128 + ic] = (_Float16)(w[oc * 3200 + i] * scale);
    }
    if (threadIdx.x == 0) bf[oc] = (b[oc] - m[oc]) * scale + beta[oc];
}

// (8,256e,256o) -> per-head transpose (8,256o,256e), f16
__global__ void k_transpose_heads(const float* __restrict__ W, _Float16* __restrict__ Wt) {
    const int total = 8 * 256 * 256;
    for (int i = blockIdx.x * blockDim.x + threadIdx.x; i < total;
         i += gridDim.x * blockDim.x) {
        int h = i >> 16, rem = i & 65535;
        int e = rem >> 8, o = rem & 255;
        Wt[(h << 16) + o * 256 + e] = (_Float16)W[i];
    }
}

// W: R x C row-major f32  ->  Wt: C x R f16
__global__ void k_transpose_h(const float* __restrict__ W, _Float16* __restrict__ Wt,
                              int R, int C) {
    int total = R * C;
    for (int i = blockIdx.x * blockDim.x + threadIdx.x; i < total;
         i += gridDim.x * blockDim.x) {
        int r = i / C, c = i - r * C;
        Wt[c * R + r] = (_Float16)W[i];
    }
}

// ------------------------- fused conv stack per chunk ----------------------
// One block per chunk (22528 blocks).  conv1 scalar, conv2/conv3 WMMA,
// activations never leave LDS until the 256-wide feature vector.

__global__ __launch_bounds__(256) void k_convstack(
    const float* __restrict__ a1, const float* __restrict__ a2,
    const float* __restrict__ w1f, const float* __restrict__ b1f,
    const _Float16* __restrict__ w2r, const float* __restrict__ b2f,
    const _Float16* __restrict__ w3r, const float* __restrict__ b3f,
    _Float16* __restrict__ feat_h) {
    __shared__ __align__(16) float xs[80];
    __shared__ __align__(16) float w1s[64 * 26];
    __shared__ __align__(16) float b1s[64];
    __shared__ __align__(16) float b2s[128];
    __shared__ __align__(16) float b3s[256];
    __shared__ __align__(16) _Float16 y1t[64 * 64];   // [p(+k)][ic], ld 64
    __shared__ __align__(16) _Float16 y2t[32 * 128];  // [p][oc],    ld 128

    const int t = threadIdx.x;
    const int lane = t & 31;
    const int wave = t >> 5;
    const int g = blockIdx.x;  // chunk id 0..22527
    const int icoff = (lane >> 4) << 4;  // 0 | 16

    // ---- gather chunk + stage conv weights/biases; clear y1t pad rows ----
    {
        int which = (g >= 11264) ? 1 : 0;
        int idx = g - which * 11264;
        int b = idx / 22, c = idx - b * 22;
        const float* src = which ? a2 : a1;
        if (t < 75) xs[t] = src[b * 600 + c * 25 + t];
        for (int i = t; i < 64 * 26; i += 256) w1s[i] = w1f[i];
        if (t < 64) b1s[t] = b1f[t];
        if (t < 128) b2s[t] = b2f[t];
        b3s[t] = b3f[t];
        // rows 50..63 of y1t feed only discarded (padding) output columns,
        // but keep them finite:
        u32x4 z = {0u, 0u, 0u, 0u};
        ((u32x4*)y1t)[t] = z;                 // halves [0, 2048)
        ((u32x4*)y1t)[256 + t] = z;           // halves [2048, 4096)
    }
    __syncthreads();

    // ---- conv1 (in-ch 1, k=26) + folded bias + SiLU -> y1t[p][ic] ----
    {
        const int oc = t & 63;
        const int pb = t >> 6;  // 0..3
        const float bb = b1s[oc];
        const float* wr = &w1s[oc * 26];
#pragma unroll
        for (int j = 0; j < 13; ++j) {
            int p = pb + j * 4;
            if (p < 50) {
                float s = bb;
#pragma unroll
                for (int k = 0; k < 26; ++k) s += wr[k] * xs[p + k];
                y1t[p * 64 + oc] = (_Float16)silu(s);
            }
        }
    }
    __syncthreads();

    // ---- conv2 GEMM: [128 oc] x [32 pos(25 valid)], K = 26*64 = 1664 ----
    // kk = k*64 + ic : each 32-wide K-step sits inside one tap k, so the
    // B fragment is a contiguous 16-half run of y1t[p + k][icbase ..].
    v8f acc0 = zero8();
    v8f acc1 = zero8();
    const int ocb = wave * 16;  // each wave owns one 16-row oc tile
    for (int ks = 0; ks < 52; ++ks) {
        const int k = ks >> 1;
        const int icbase = ((ks & 1) << 5) + icoff;
        __builtin_prefetch(w2r + (ocb + (lane & 15)) * 1664 + (ks + 1) * 32, 0, 3);
        v16h a = load_a_frag(w2r, 1664, ocb + (lane & 15), ks * 32, lane);
        v16h b0 = load_b_run(y1t + ((lane & 15) + k) * 64 + icbase);
        v16h b1 = load_b_run(y1t + ((lane & 15) + 16 + k) * 64 + icbase);
        acc0 = wmma16(a, b0, acc0);
        acc1 = wmma16(a, b1, acc1);
    }
    __syncthreads();  // everyone done reading y1t before epilogue writes
    // bias + SiLU -> y2t[p][oc] (zero rows p >= 25)
#pragma unroll
    for (int i = 0; i < 8; ++i) {
        int oc = ocb + i + ((lane >> 4) << 3);
        int p0 = lane & 15;
        int p1 = p0 + 16;
        float bb = b2s[oc];
        y2t[p0 * 128 + oc] = (_Float16)silu(acc0[i] + bb);
        y2t[p1 * 128 + oc] =
            (p1 < 25) ? (_Float16)silu(acc1[i] + bb) : (_Float16)0.f;
    }
    __syncthreads();

    // ---- conv3 GEMV via WMMA: [256 oc] x [16 (col 0 valid)], K = 3200 ----
    // kk = k*128 + ic : B fragment = contiguous run of y2t[k][icbase ..].
    v8f c0 = zero8();
    v8f c1 = zero8();
    for (int ks = 0; ks < 100; ++ks) {
        FragH bf;
        if ((lane & 15) == 0) {  // only output column 0 carries data
            const _Float16* p =
                y2t + (ks >> 2) * 128 + ((ks & 3) << 5) + icoff;
            bf.q[0] = *(const u32x4*)p;
            bf.q[1] = *(const u32x4*)(p + 8);
        } else {
#pragma unroll
            for (int i = 0; i < 8; ++i) bf.w[i] = 0u;
        }
        __builtin_prefetch(w3r + (wave * 32 + (lane & 15)) * 3200 + (ks + 1) * 32, 0, 3);
        v16h a0 = load_a_frag(w3r, 3200, wave * 32 + (lane & 15), ks * 32, lane);
        v16h a1 = load_a_frag(w3r, 3200, wave * 32 + 16 + (lane & 15), ks * 32, lane);
        c0 = wmma16(a0, bf.h, c0);
        c1 = wmma16(a1, bf.h, c1);
    }
    if ((lane & 15) == 0) {
#pragma unroll
        for (int i = 0; i < 8; ++i) {
            int oc = wave * 32 + i + ((lane >> 4) << 3);
            float v0 = c0[i] + b3s[oc];
            feat_h[(size_t)g * 256 + oc] = (_Float16)silu(v0);
            int oc2 = oc + 16;
            float v1 = c1[i] + b3s[oc2];
            feat_h[(size_t)g * 256 + oc2] = (_Float16)silu(v1);
        }
    }
}

// --------------------------- fused attention head --------------------------
// One block per (head h, batch b).  QKV -> QK^T -> softmax -> A*V -> O.

__global__ __launch_bounds__(256) void k_attn(
    const _Float16* __restrict__ feat_h,
    const _Float16* __restrict__ wqt, const _Float16* __restrict__ wkt,
    const _Float16* __restrict__ wvt,
    const float* __restrict__ bq, const float* __restrict__ bk,
    const float* __restrict__ bv,
    _Float16* __restrict__ Oh) {
    __shared__ __align__(16) _Float16 Qs[32 * 264];  // s-major, ld 264
    __shared__ __align__(16) _Float16 Ks[32 * 264];  // t-major, ld 264
    __shared__ __align__(16) _Float16 Vt[256 * 40];  // o-major (transposed), ld 40
    __shared__ __align__(16) float sc[32 * 32];      // QK^T scores
    __shared__ __align__(16) _Float16 Ah[32 * 32];   // softmax(A), zero padded

    const int t = threadIdx.x;
    const int lane = t & 31;
    const int wave = t >> 5;
    const int h = blockIdx.x;
    const int b = blockIdx.y;
    const _Float16* fb = feat_h + (size_t)b * 22 * 256;

    // ---- Q,K,V projections: 3 mats x (2 M-tiles x 16 N-tiles) = 96 tiles ----
    for (int tid = wave; tid < 96; tid += 8) {
        int m = tid >> 5;  // 0=Q 1=K 2=V
        int tt = tid & 31;
        int mt = tt >> 4, nt = tt & 15;
        const _Float16* W =
            (m == 0 ? wqt : (m == 1 ? wkt : wvt)) + ((size_t)h << 16);
        const float* bias = (m == 0 ? bq : (m == 1 ? bk : bv)) + h * 256;
        v8f acc = zero8();
        for (int ks = 0; ks < 8; ++ks) {
            int s = mt * 16 + (lane & 15);
            v16h a = (s < 22) ? load_a_frag(fb, 256, s, ks * 32, lane)
                              : frag_zero();
            v16h bb = load_b_frag(W, 256, nt * 16 + (lane & 15), ks * 32, lane);
            acc = wmma16(a, bb, acc);
        }
        int col = nt * 16 + (lane & 15);
        float bo = bias[col];
#pragma unroll
        for (int i = 0; i < 8; ++i) {
            int row = mt * 16 + i + ((lane >> 4) << 3);
            float v = acc[i] + bo;
            if (m == 0)       Qs[row * 264 + col] = (_Float16)v;
            else if (m == 1)  Ks[row * 264 + col] = (_Float16)v;
            else              Vt[col * 40 + row] = (_Float16)v;  // transposed
        }
    }
    __syncthreads();

    // ---- scores = Q K^T : 2x2 tiles, K = 256 ----
    if (wave < 4) {
        int mt = wave >> 1, nt = wave & 1;
        v8f acc = zero8();
        for (int ks = 0; ks < 8; ++ks) {
            v16h a = load_a_frag(Qs, 264, mt * 16 + (lane & 15), ks * 32, lane);
            v16h bb = load_b_frag(Ks, 264, nt * 16 + (lane & 15), ks * 32, lane);
            acc = wmma16(a, bb, acc);
        }
        int colt = nt * 16 + (lane & 15);
#pragma unroll
        for (int i = 0; i < 8; ++i)
            sc[(mt * 16 + i + ((lane >> 4) << 3)) * 32 + colt] = acc[i];
    }
    __syncthreads();

    // ---- softmax over t (22 valid), write f16 A with zero padding ----
    if (t < 32) {
        int r = t;
        if (r < 22) {
            float mx = -1e30f;
            for (int j = 0; j < 22; ++j) mx = fmaxf(mx, sc[r * 32 + j]);
            float sum = 0.f;
            for (int j = 0; j < 22; ++j) sum += __expf(sc[r * 32 + j] - mx);
            float inv = 1.f / sum;
            for (int j = 0; j < 32; ++j)
                Ah[r * 32 + j] = (j < 22)
                    ? (_Float16)(__expf(sc[r * 32 + j] - mx) * inv)
                    : (_Float16)0.f;
        } else {
            for (int j = 0; j < 32; ++j) Ah[r * 32 + j] = (_Float16)0.f;
        }
    }
    __syncthreads();

    // ---- O = A V : 2 M-tiles x 16 N-tiles, single K step (32, 22 valid) ----
    for (int tid = wave; tid < 32; tid += 8) {
        int mt = tid >> 4, nt = tid & 15;
        v16h a = load_a_frag(Ah, 32, mt * 16 + (lane & 15), 0, lane);
        v16h bb = load_b_frag(Vt, 40, nt * 16 + (lane & 15), 0, lane);
        v8f acc = wmma16(a, bb, zero8());
        int col = nt * 16 + (lane & 15);
#pragma unroll
        for (int i = 0; i < 8; ++i) {
            int s = mt * 16 + i + ((lane >> 4) << 3);
            if (s < 22)
                Oh[(size_t)(b * 22 + s) * 2048 + h * 256 + col] =
                    (_Float16)acc[i];
        }
    }
}

// ------------------------- generic f16 WMMA GEMM ---------------------------
// C[M x N](f32) = A[M x K](f16, row-major) * Bt[N x K](f16) + bias[N]
// Block computes a 128 x 32 tile; grid = (M/128, N/32).

__global__ __launch_bounds__(256) void k_gemm_f16(
    const _Float16* __restrict__ A, const _Float16* __restrict__ Bt,
    const float* __restrict__ bias, float* __restrict__ C, int Kdim, int N) {
    const int t = threadIdx.x;
    const int lane = t & 31;
    const int wave = t >> 5;
    const int row0 = blockIdx.x * 128 + wave * 16;
    const int col0 = blockIdx.y * 32;

    v8f acc0 = zero8();
    v8f acc1 = zero8();
    for (int ks = 0; ks < Kdim; ks += 32) {
        __builtin_prefetch(A + (row0 + (lane & 15)) * Kdim + ks + 32, 0, 3);
        v16h a = load_a_frag(A, Kdim, row0 + (lane & 15), ks, lane);
        v16h b0 = load_b_frag(Bt, Kdim, col0 + (lane & 15), ks, lane);
        v16h b1 = load_b_frag(Bt, Kdim, col0 + 16 + (lane & 15), ks, lane);
        acc0 = wmma16(a, b0, acc0);
        acc1 = wmma16(a, b1, acc1);
    }
    int c0 = col0 + (lane & 15);
    int c1 = c0 + 16;
    float bb0 = bias[c0], bb1 = bias[c1];
#pragma unroll
    for (int i = 0; i < 8; ++i) {
        int r = row0 + i + ((lane >> 4) << 3);
        C[(size_t)r * N + c0] = acc0[i] + bb0;
        C[(size_t)r * N + c1] = acc1[i] + bb1;
    }
}

// ------------------------------- mean pool ---------------------------------

__global__ void k_pool(const float* __restrict__ outf, _Float16* __restrict__ pooledh) {
    int b = blockIdx.x;   // 1024
    int o = threadIdx.x;  // 256
    float s = 0.f;
    for (int j = 0; j < 22; ++j) s += outf[(size_t)(b * 22 + j) * 256 + o];
    pooledh[b * 256 + o] = (_Float16)(s * (1.f / 22.f));
}

// ------------------------------- launcher ----------------------------------

extern "C" void kernel_launch(void* const* d_in, const int* in_sizes, int n_in,
                              void* d_out, int out_size, void* d_ws, size_t ws_size,
                              hipStream_t stream) {
    const float* a1  = (const float*)d_in[0];
    const float* a2  = (const float*)d_in[1];
    const float* cw1 = (const float*)d_in[2];
    const float* cb1 = (const float*)d_in[3];
    const float* g1  = (const float*)d_in[4];
    const float* bt1 = (const float*)d_in[5];
    const float* m1  = (const float*)d_in[6];
    const float* v1  = (const float*)d_in[7];
    const float* cw2 = (const float*)d_in[8];
    const float* cb2 = (const float*)d_in[9];
    const float* g2  = (const float*)d_in[10];
    const float* bt2 = (const float*)d_in[11];
    const float* m2  = (const float*)d_in[12];
    const float* v2  = (const float*)d_in[13];
    const float* cw3 = (const float*)d_in[14];
    const float* cb3 = (const float*)d_in[15];
    const float* g3  = (const float*)d_in[16];
    const float* bt3 = (const float*)d_in[17];
    const float* m3  = (const float*)d_in[18];
    const float* v3  = (const float*)d_in[19];
    const float* Wq  = (const float*)d_in[20];
    const float* bq  = (const float*)d_in[21];
    const float* Wk  = (const float*)d_in[22];
    const float* bk  = (const float*)d_in[23];
    const float* Wv  = (const float*)d_in[24];
    const float* bv  = (const float*)d_in[25];
    const float* fca_w = (const float*)d_in[26];
    const float* fca_b = (const float*)d_in[27];
    const float* fc_w  = (const float*)d_in[28];
    const float* fc_b  = (const float*)d_in[29];
    float* out = (float*)d_out;

    // Workspace layout (~135 MB total), 256B aligned slices.
    char* ws = (char*)d_ws;
    size_t off = 0;
    auto alloc = [&](size_t bytes) -> char* {
        char* p = ws + off;
        off += (bytes + 255) & ~(size_t)255;
        return p;
    };
    float*    w1f   = (float*)alloc(64 * 26 * 4);
    float*    b1f   = (float*)alloc(64 * 4);
    _Float16* w2r   = (_Float16*)alloc((size_t)128 * 1664 * 2);
    float*    b2f   = (float*)alloc(128 * 4);
    _Float16* w3r   = (_Float16*)alloc((size_t)256 * 3200 * 2);
    float*    b3f   = (float*)alloc(256 * 4);
    _Float16* wqt   = (_Float16*)alloc((size_t)8 * 256 * 256 * 2);
    _Float16* wkt   = (_Float16*)alloc((size_t)8 * 256 * 256 * 2);
    _Float16* wvt   = (_Float16*)alloc((size_t)8 * 256 * 256 * 2);
    _Float16* fcawt = (_Float16*)alloc((size_t)2048 * 256 * 2);
    _Float16* fcwt  = (_Float16*)alloc((size_t)256 * 256 * 2);
    _Float16* feath = (_Float16*)alloc((size_t)22528 * 256 * 2);
    _Float16* Oh    = (_Float16*)alloc((size_t)22528 * 2048 * 2);
    float*    outf  = (float*)alloc((size_t)22528 * 256 * 4);
    _Float16* poolh = (_Float16*)alloc((size_t)1024 * 256 * 2);

    // prep: BN folding + f16 weight transposes / reorders
    k_fold_conv1<<<1, 64, 0, stream>>>(cw1, cb1, g1, bt1, m1, v1, w1f, b1f);
    k_fold_conv2<<<128, 256, 0, stream>>>(cw2, cb2, g2, bt2, m2, v2, w2r, b2f);
    k_fold_conv3<<<256, 256, 0, stream>>>(cw3, cb3, g3, bt3, m3, v3, w3r, b3f);
    k_transpose_heads<<<512, 256, 0, stream>>>(Wq, wqt);
    k_transpose_heads<<<512, 256, 0, stream>>>(Wk, wkt);
    k_transpose_heads<<<512, 256, 0, stream>>>(Wv, wvt);
    k_transpose_h<<<512, 256, 0, stream>>>(fca_w, fcawt, 2048, 256);
    k_transpose_h<<<64, 256, 0, stream>>>(fc_w, fcwt, 256, 256);

    // fused conv stack: one block per chunk
    k_convstack<<<22528, 256, 0, stream>>>(a1, a2, w1f, b1f, w2r, b2f, w3r, b3f,
                                           feath);
    // attention: one block per (head, batch-row)
    k_attn<<<dim3(8, 1024), 256, 0, stream>>>(feath, wqt, wkt, wvt, bq, bk, bv, Oh);
    // fca: (22528 x 256) = O(22528 x 2048) @ fcawt
    k_gemm_f16<<<dim3(176, 8), 256, 0, stream>>>(Oh, fcawt, fca_b, outf, 2048, 256);
    // mean over the 22 chunk positions
    k_pool<<<1024, 256, 0, stream>>>(outf, poolh);
    // final fc: (1024 x 256) = pooled @ fcwt  -> d_out
    k_gemm_f16<<<dim3(8, 8), 256, 0, stream>>>(poolh, fcwt, fc_b, out, 256, 256);

    (void)in_sizes; (void)n_in; (void)out_size; (void)ws_size;
}
